// ACTP_40252433498661
// MI455X (gfx1250) — matrix operand order repaired and checked by reading, not compiled, not run on hardware
//
#include <hip/hip_runtime.h>
#include <hip/hip_bf16.h>

// ---------------------------------------------------------------------------
// Problem constants (match the JAX reference)
// ---------------------------------------------------------------------------
#define BB   512
#define DD   64
#define AA   12
#define HH   512
#define G4H  2048
#define KT1  18            // K=576  (D + H)        LSTM1
#define KT2  35            // K=1120 (H + 96 + H)   LSTM2
#define KT3  18            // K=576  (H + D)        MLP1
#define KT4  16            // K=512  (H)            MLP2
#define CTX  10
#define NSTEP 100

typedef __attribute__((ext_vector_type(16))) __bf16        v16bf;
typedef __attribute__((ext_vector_type(8)))  float         v8f;
typedef __attribute__((ext_vector_type(8)))  unsigned int  v8u;

static __device__ __forceinline__ unsigned short f2bf(float f) {
    unsigned int u = __builtin_bit_cast(unsigned int, f);
    unsigned int r = (u + 0x7FFFu + ((u >> 16) & 1u)) >> 16;   // RNE
    return (unsigned short)r;
}

// A-fragment K map (16-bit A 16x32, ISA 7.12.2): element e of lane L covers
//   K = e + (e&8) + ((L>>4)<<3)
static __device__ __forceinline__ int kmapA(int lane, int e) {
    return e + (e & 8) + ((lane >> 4) << 3);
}

// Scatter one bf16 element (row m, col k) into a packed A-fragment buffer
// with KtD K-tiles (inverse of the fragment map above).
static __device__ __forceinline__ void storeA(unsigned short* __restrict__ p,
                                              int KtD, int k, int m,
                                              unsigned short v) {
    int kin   = k & 31;
    int kt2   = k >> 5;
    int half  = (kin >> 3) & 1;
    int e     = (kin & 7) + ((kin & 16) ? 8 : 0);
    int lanep = (half << 4) | (m & 15);
    p[(((size_t)(m >> 4) * KtD + kt2) * 32 + lanep) * 16 + e] = v;
}

// ---------------------------------------------------------------------------
// Weight pre-pack: fp32 W (rows = output cols, up to two K-segments) -> bf16
// B-fragments. grouped=1 reorders gate rows so packed col n' = jt*64+g*16+c
// maps to original row g*512 + jt*16 + c  ([i f g o] interleaved per j-tile).
// ---------------------------------------------------------------------------
__global__ void pack_w_kernel(const float* __restrict__ s0, int k0len,
                              const float* __restrict__ s1, int k1len,
                              unsigned short* __restrict__ frag,
                              int Ntiles, int Ktiles, int grouped) {
    int gid  = blockIdx.x * blockDim.x + threadIdx.x;
    int lane = gid & 31;
    int tile = gid >> 5;
    if (tile >= Ntiles * Ktiles) return;
    int nt = tile / Ktiles, kt = tile % Ktiles;
    int n  = nt * 16 + (lane & 15);
    if (grouped) n = ((n >> 4) & 3) * 512 + (n >> 6) * 16 + (n & 15);
    int kstart = kt * 32 + ((lane >> 4) << 3);
    const float* src; int stride, kk;
    if (kstart < k0len) { src = s0; stride = k0len; kk = kstart; }
    else                { src = s1; stride = k1len; kk = kstart - k0len; }
    const float* row = src + (size_t)n * stride + kk;
    unsigned short* dst = frag + ((size_t)tile * 32 + lane) * 16;
#pragma unroll
    for (int e = 0; e < 16; ++e) dst[e] = f2bf(row[e + (e & 8)]);
}

__global__ void pack_bias_group_kernel(const float* __restrict__ ba,
                                       const float* __restrict__ bb,
                                       float* __restrict__ o) {
    int n = blockIdx.x * blockDim.x + threadIdx.x;
    if (n >= G4H) return;
    int orig = ((n >> 4) & 3) * 512 + (n >> 6) * 16 + (n & 15);
    o[n] = ba[orig] + bb[orig];
}

// ---------------------------------------------------------------------------
// WMMA GEMM, fully pre-packed operands. Block = 8 waves over M (256 rows);
// wave tile = 32M x (NT*16)N; KT compile-time -> all frag loads are
// base + immediate offset, bases advance 1KB/iteration.
// omode: 1 = tanh(+bias) fp32 | 2 = tanh(+bias) packed-A bf16
//        3 = fused LSTM cell (NT==8, gate-grouped weights): h written as
//            packed A-fragments into TWO destination buffers, c updated.
// ---------------------------------------------------------------------------
template <int NT, int KT>
__global__ __launch_bounds__(256) void wmma_gemm_kernel(
    const unsigned short* __restrict__ Af, const unsigned short* __restrict__ Wf,
    const float* __restrict__ bias, float* __restrict__ outF, int N, int omode,
    unsigned short* __restrict__ pD0, int ktD0, int koff0,
    unsigned short* __restrict__ pD1, int ktD1, int koff1,
    float* __restrict__ cbuf) {
    const int lane = threadIdx.x & 31;
    const int wave = threadIdx.x >> 5;
    const int mtileBase = blockIdx.y * 16 + wave * 2;
    const int ntileBase = blockIdx.x * NT;

    const v8f vzero = {0.f, 0.f, 0.f, 0.f, 0.f, 0.f, 0.f, 0.f};
    v8f acc[2][NT];
#pragma unroll
    for (int mt = 0; mt < 2; ++mt)
#pragma unroll
        for (int nt = 0; nt < NT; ++nt) acc[mt][nt] = vzero;

    const v8u* ap = reinterpret_cast<const v8u*>(Af) + (size_t)mtileBase * KT * 32 + lane;
    const v8u* bp = reinterpret_cast<const v8u*>(Wf) + (size_t)ntileBase * KT * 32 + lane;

#pragma unroll 1
    for (int kt = 0; kt < KT; ++kt) {
        v8u aF[2], bF[NT];
#pragma unroll
        for (int mt = 0; mt < 2; ++mt) aF[mt] = ap[(size_t)mt * KT * 32];
#pragma unroll
        for (int nt = 0; nt < NT; ++nt) bF[nt] = bp[(size_t)nt * KT * 32];
        __builtin_prefetch(ap + 32, 0, 0);
        __builtin_prefetch(bp + 32, 0, 0);
        ap += 32; bp += 32;
#pragma unroll
        for (int mt = 0; mt < 2; ++mt)
#pragma unroll
            for (int nt = 0; nt < NT; ++nt)
                acc[mt][nt] = __builtin_amdgcn_wmma_f32_16x16x32_bf16(
                    false, __builtin_bit_cast(v16bf, aF[mt]),
                    false, __builtin_bit_cast(v16bf, bF[nt]),
                    (short)0, acc[mt][nt], false, false);
    }

    // ---- epilogues ----------------------------------------------------
    if constexpr (NT == 8) {
        if (omode == 3) {                       // fused LSTM cell
            const int jt0 = ntileBase >> 2;     // j-tile base (grouped layout)
#pragma unroll
            for (int q = 0; q < 2; ++q) {
                int cix = lane & 15;
                int j   = (jt0 + q) * 16 + cix;
                float bi = bias[(jt0 + q) * 64 + cix];
                float bf = bias[(jt0 + q) * 64 + 16 + cix];
                float bg = bias[(jt0 + q) * 64 + 32 + cix];
                float bo = bias[(jt0 + q) * 64 + 48 + cix];
#pragma unroll
                for (int mt = 0; mt < 2; ++mt) {
                    int mrow = (mtileBase + mt) * 16 + ((lane >> 4) << 3);
#pragma unroll
                    for (int r = 0; r < 8; ++r) {
                        int m = mrow + r;
                        float gi = acc[mt][q * 4 + 0][r] + bi;
                        float gf = acc[mt][q * 4 + 1][r] + bf;
                        float gg = acc[mt][q * 4 + 2][r] + bg;
                        float go = acc[mt][q * 4 + 3][r] + bo;
                        float si = 1.f / (1.f + __expf(-gi));
                        float sf = 1.f / (1.f + __expf(-gf));
                        float so = 1.f / (1.f + __expf(-go));
                        size_t ci = (size_t)m * HH + j;
                        float cn = sf * cbuf[ci] + si * tanhf(gg);
                        cbuf[ci] = cn;
                        unsigned short hb = f2bf(so * tanhf(cn));
                        storeA(pD0, ktD0, koff0 + j, m, hb);
                        storeA(pD1, ktD1, koff1 + j, m, hb);
                    }
                }
            }
            return;
        }
    }
#pragma unroll
    for (int nt = 0; nt < NT; ++nt) {
        int n = (ntileBase + nt) * 16 + (lane & 15);
        float bvv = bias[n];
#pragma unroll
        for (int mt = 0; mt < 2; ++mt) {
            int mrow = (mtileBase + mt) * 16 + ((lane >> 4) << 3);
#pragma unroll
            for (int r = 0; r < 8; ++r) {
                float v = tanhf(acc[mt][nt][r] + bvv);
                int m = mrow + r;
                if (omode == 1) outF[(size_t)m * N + n] = v;
                else            storeA(pD0, ktD0, koff0 + n, m, f2bf(v));
            }
        }
    }
}

// ---------------------------------------------------------------------------
// Builders: inp1 appears in X1 at k in [0,64) and in X3 at k in [512,576) —
// identical packed tiles, so one kernel fills both. Tiled actions fill
// Xf2 kt 16..18 (k in [512,608)).
// ---------------------------------------------------------------------------
__global__ void build_inp_kernel(unsigned short* __restrict__ Xf1,
                                 unsigned short* __restrict__ Xf3,
                                 const float* __restrict__ tac_t,
                                 const float* __restrict__ prev, int use_tac) {
    int gid = blockIdx.x * blockDim.x + threadIdx.x;
    int lane = gid & 31, tile = gid >> 5;
    if (tile >= 32 * 2) return;
    int mtile = tile >> 1, kt = tile & 1;
    int b = mtile * 16 + (lane & 15);
    const float* inp1 = use_tac ? tac_t : prev;
    unsigned short vals[16];
#pragma unroll
    for (int e = 0; e < 16; ++e)
        vals[e] = f2bf(inp1[b * DD + kt * 32 + kmapA(lane, e)]);
    unsigned short* d1 = Xf1 + (((size_t)mtile * KT1 + kt) * 32 + lane) * 16;
    unsigned short* d3 = Xf3 + (((size_t)mtile * KT3 + 16 + kt) * 32 + lane) * 16;
#pragma unroll
    for (int e = 0; e < 16; ++e) { d1[e] = vals[e]; d3[e] = vals[e]; }
}

__global__ void build_tiled_kernel(unsigned short* __restrict__ Xf2,
                                   const float* __restrict__ act,
                                   const float* __restrict__ st0) {
    int gid = blockIdx.x * blockDim.x + threadIdx.x;
    int lane = gid & 31, tile = gid >> 5;
    if (tile >= 32 * 3) return;
    int mtile = tile / 3, kt = 16 + tile % 3;
    int b = mtile * 16 + (lane & 15);
    unsigned short* dst = Xf2 + (((size_t)mtile * KT2 + kt) * 32 + lane) * 16;
#pragma unroll
    for (int e = 0; e < 16; ++e) {
        int k = kt * 32 + kmapA(lane, e);          // 512..607
        int j = (k - HH) % 24;                     // tiled = [act, state0] * 4
        float v = (j < AA) ? act[b * AA + j] : st0[b * AA + (j - AA)];
        dst[e] = f2bf(v);
    }
}

// ---------------------------------------------------------------------------
// Per-step epilogue: copy out4 to d_out (if t>=CTX-1) and accumulate MAE.
// Single workgroup -> deterministic accumulation order.
// ---------------------------------------------------------------------------
__global__ void post_step_kernel(const float* __restrict__ out4,
                                 const float* __restrict__ tacnext,
                                 float* __restrict__ mae_accum,
                                 float* __restrict__ outdst, int do_out) {
    __shared__ float red[1024];
    int tid = threadIdx.x;
    float s = 0.f;
    for (int i = tid; i < BB * DD; i += 1024) {
        float v = out4[i];
        s += fabsf(v - tacnext[i]);
        if (do_out) outdst[i] = v;
    }
    red[tid] = s;
    __syncthreads();
    for (int off = 512; off > 0; off >>= 1) {
        if (tid < off) red[tid] += red[tid + off];
        __syncthreads();
    }
    if (tid == 0) *mae_accum += red[0] * (1.0f / ((float)BB * (float)DD * (float)NSTEP));
}

__global__ void zero_kernel(unsigned int* p, size_t n) {
    size_t i = (size_t)blockIdx.x * blockDim.x + threadIdx.x;
    if (i < n) p[i] = 0u;
}

// ---------------------------------------------------------------------------
extern "C" void kernel_launch(void* const* d_in, const int* in_sizes, int n_in,
                              void* d_out, int out_size, void* d_ws, size_t ws_size,
                              hipStream_t stream) {
    const float* tact = (const float*)d_in[0];   // (T,B,D)
    const float* acts = (const float*)d_in[1];   // (T,B,A)
    const float* Wih1 = (const float*)d_in[2];
    const float* Whh1 = (const float*)d_in[3];
    const float* bih1 = (const float*)d_in[4];
    const float* bhh1 = (const float*)d_in[5];
    const float* Wih2 = (const float*)d_in[6];
    const float* Whh2 = (const float*)d_in[7];
    const float* bih2 = (const float*)d_in[8];
    const float* bhh2 = (const float*)d_in[9];
    const float* W1   = (const float*)d_in[10];
    const float* b1   = (const float*)d_in[11];
    const float* W2   = (const float*)d_in[12];
    const float* b2   = (const float*)d_in[13];
    float* outp = (float*)d_out;                 // [0]=mae, [1..]=outputs(91,512,64)

    char* base = (char*)d_ws;
    size_t off = 0;
    auto carve = [&](size_t bytes) -> char* {
        char* p = base + off;
        off += (bytes + 255) & ~(size_t)255;
        return p;
    };
    const int NT_G = G4H / 16, NT_M1 = HH / 16, NT_M2 = DD / 16;
    const int MT = BB / 16;
    unsigned short* Wc1f = (unsigned short*)carve((size_t)NT_G  * KT1 * 1024);
    unsigned short* Wc2f = (unsigned short*)carve((size_t)NT_G  * KT2 * 1024);
    unsigned short* W1f  = (unsigned short*)carve((size_t)NT_M1 * KT3 * 1024);
    unsigned short* W2f  = (unsigned short*)carve((size_t)NT_M2 * KT4 * 1024);
    unsigned short* Xf1[2], *Xf2[2];
    Xf1[0] = (unsigned short*)carve((size_t)MT * KT1 * 1024);
    Xf1[1] = (unsigned short*)carve((size_t)MT * KT1 * 1024);
    Xf2[0] = (unsigned short*)carve((size_t)MT * KT2 * 1024);
    Xf2[1] = (unsigned short*)carve((size_t)MT * KT2 * 1024);
    unsigned short* Xf3 = (unsigned short*)carve((size_t)MT * KT3 * 1024);
    unsigned short* Xf4 = (unsigned short*)carve((size_t)MT * KT4 * 1024);
    float* bc1  = (float*)carve(G4H * 4);
    float* bc2  = (float*)carve(G4H * 4);
    float* c1   = (float*)carve((size_t)BB * HH * 4);
    float* c2   = (float*)carve((size_t)BB * HH * 4);
    float* out4 = (float*)carve((size_t)BB * DD * 4);

    // ---- one-time (per launch) packing + zero-init ----
    {
        int th, bl;
        th = NT_G * KT1 * 32;  bl = (th + 255) / 256;
        pack_w_kernel<<<bl, 256, 0, stream>>>(Wih1, DD, Whh1, HH, Wc1f, NT_G, KT1, 1);
        th = NT_G * KT2 * 32;  bl = (th + 255) / 256;
        pack_w_kernel<<<bl, 256, 0, stream>>>(Wih2, HH + 96, Whh2, HH, Wc2f, NT_G, KT2, 1);
        th = NT_M1 * KT3 * 32; bl = (th + 255) / 256;
        pack_w_kernel<<<bl, 256, 0, stream>>>(W1, HH + DD, W1, 0, W1f, NT_M1, KT3, 0);
        th = NT_M2 * KT4 * 32; bl = (th + 255) / 256;
        pack_w_kernel<<<bl, 256, 0, stream>>>(W2, HH, W2, 0, W2f, NT_M2, KT4, 0);
        pack_bias_group_kernel<<<(G4H + 255) / 256, 256, 0, stream>>>(bih1, bhh1, bc1);
        pack_bias_group_kernel<<<(G4H + 255) / 256, 256, 0, stream>>>(bih2, bhh2, bc2);
        auto z = [&](void* p, size_t bytes) {
            size_t n = bytes / 4;
            zero_kernel<<<(unsigned)((n + 255) / 256), 256, 0, stream>>>((unsigned int*)p, n);
        };
        z(c1, (size_t)BB * HH * 4);  z(c2, (size_t)BB * HH * 4);
        z(out4, (size_t)BB * DD * 4);
        z(Xf1[0], (size_t)MT * KT1 * 1024); z(Xf1[1], (size_t)MT * KT1 * 1024);
        z(Xf2[0], (size_t)MT * KT2 * 1024); z(Xf2[1], (size_t)MT * KT2 * 1024);
        z(Xf3, (size_t)MT * KT3 * 1024);
        zero_kernel<<<1, 32, 0, stream>>>((unsigned int*)outp, 1);   // mae
    }

    // ---- the 100 sequential steps ----
    const float* st0 = acts;                      // actions[0] = state0
    for (int t = 0; t < NSTEP; ++t) {
        const float* tac_t = tact + (size_t)t * BB * DD;
        const float* tac_n = tact + (size_t)(t + 1) * BB * DD;
        const float* act_t = acts + (size_t)(t + 1) * BB * AA;
        int cur = t & 1, nxt = cur ^ 1;
        int use_tac = (t < CTX);
        int do_out  = (t >= CTX - 1);
        float* odst = outp + 1 + (size_t)(do_out ? (t - (CTX - 1)) : 0) * BB * DD;

        build_inp_kernel<<<8, 256, 0, stream>>>(Xf1[cur], Xf3, tac_t, out4, use_tac);
        build_tiled_kernel<<<12, 256, 0, stream>>>(Xf2[cur], act_t, st0);

        // LSTM1: gates GEMM + fused cell; h1 -> Xf2[cur] (k=0..511) and
        //        next step's Xf1 (k=64..575)
        wmma_gemm_kernel<8, KT1><<<dim3(G4H / 128, BB / 256), 256, 0, stream>>>(
            Xf1[cur], Wc1f, bc1, nullptr, G4H, 3,
            Xf2[cur], KT2, 0, Xf1[nxt], KT1, DD, c1);

        // LSTM2: gates GEMM + fused cell; h2 -> next step's Xf2 (k=608..1119)
        //        and Xf3 (k=0..511)
        wmma_gemm_kernel<8, KT2><<<dim3(G4H / 128, BB / 256), 256, 0, stream>>>(
            Xf2[cur], Wc2f, bc2, nullptr, G4H, 3,
            Xf2[nxt], KT2, HH + 96, Xf3, KT3, 0, c2);

        // MLP1: tanh output written directly as packed A-fragments for MLP2
        wmma_gemm_kernel<8, KT3><<<dim3(HH / 128, BB / 256), 256, 0, stream>>>(
            Xf3, W1f, b1, nullptr, HH, 2, Xf4, KT4, 0, nullptr, 0, 0, nullptr);

        // MLP2: fp32 tanh output
        wmma_gemm_kernel<4, KT4><<<dim3(DD / 64, BB / 256), 256, 0, stream>>>(
            Xf4, W2f, b2, out4, DD, 1, nullptr, 0, 0, nullptr, 0, 0, nullptr);

        post_step_kernel<<<1, 1024, 0, stream>>>(out4, tac_n, outp, odst, do_out);
    }
}
